// VGG_67027259621653
// MI455X (gfx1250) — 16-bit and 8-bit matrix operand order repaired and checked by reading, compiled, not run on hardware
//
#include <hip/hip_runtime.h>

typedef signed char i8;
typedef __attribute__((ext_vector_type(8)))  int       v8i;
typedef __attribute__((ext_vector_type(8)))  float     v8f;
typedef __attribute__((ext_vector_type(8)))  _Float16  v8h;
typedef __attribute__((ext_vector_type(16))) _Float16  v16h;

#define BN_EPS 1e-5f

__global__ __launch_bounds__(256) void conv1_f32(const float* __restrict__ x,
                                                 const float* __restrict__ w,
                                                 float* __restrict__ out) {
  long idx = (long)blockIdx.x * 256 + threadIdx.x;
  const long total = 8L * 300 * 300 * 64;
  if (idx >= total) return;
  int co = (int)(idx & 63);
  long t = idx >> 6;
  int ow = (int)(t % 300); t /= 300;
  int oh = (int)(t % 300);
  int b  = (int)(t / 300);
  float acc = 0.f;
  #pragma unroll
  for (int kh = 0; kh < 3; ++kh) {
    int ih = oh + kh - 1;
    if (ih < 0 || ih >= 300) continue;
    #pragma unroll
    for (int kw = 0; kw < 3; ++kw) {
      int iw = ow + kw - 1;
      if (iw < 0 || iw >= 300) continue;
      #pragma unroll
      for (int ci = 0; ci < 3; ++ci)
        acc += x[(((long)b * 3 + ci) * 300 + ih) * 300 + iw] *
               w[((co * 3 + ci) * 3 + kh) * 3 + kw];
    }
  }
  out[idx] = acc;
}

__global__ __launch_bounds__(256) void binarize_w(const float* __restrict__ w,
                                                  i8* __restrict__ w8,
                                                  int Cout, int Cin) {
  long idx = (long)blockIdx.x * 256 + threadIdx.x;
  long total = (long)Cout * 9 * Cin;
  if (idx >= total) return;
  int c  = (int)(idx % Cin);
  long t = idx / Cin;
  int kk = (int)(t % 9);
  int n  = (int)(t / 9);
  float v = w[((long)n * Cin + c) * 9 + kk];
  w8[idx] = (v >= 0.f) ? (i8)1 : (i8)-1;
}

__global__ __launch_bounds__(256) void conv_bin_wmma(
    const i8* __restrict__ act, const i8* __restrict__ wgt,
    float* __restrict__ out,
    int H, int W, int OH, int OW, int Cin, int Cout, int stride, int dil) {
  const int lane = threadIdx.x & 31;
  const int wave = threadIdx.x >> 5;
  const int M = 8 * OH * OW;
  const int mTiles = (M + 15) >> 4;
  const int nTiles = Cout >> 6;
  const long tile = (long)blockIdx.x * 8 + wave;
  if (tile >= (long)mTiles * nTiles) return;

  const int nT = (int)(tile % nTiles);
  const int mT = (int)(tile / nTiles);
  const int hi = lane >> 4;
  const int r  = lane & 15;

  const int m = mT * 16 + r;
  const bool mValid = (m < M);
  const int sp = OH * OW;
  const int b  = m / sp;
  const int s  = m - b * sp;
  const int oh = s / OW, ow = s - (s / OW) * OW;

  const long Kt = 9L * Cin;
  const i8* brow = wgt + (long)(nT * 64 + r) * Kt;
  v8i acc[4];
  #pragma unroll
  for (int j = 0; j < 4; ++j) acc[j] = (v8i){0, 0, 0, 0, 0, 0, 0, 0};

  for (int kk = 0; kk < 9; ++kk) {
    const int kh = kk / 3, kw = kk - 3 * (kk / 3);
    const int ih = oh * stride + kh * dil - dil;
    const int iw = ow * stride + kw * dil - dil;
    const bool inb = mValid && (unsigned)ih < (unsigned)H && (unsigned)iw < (unsigned)W;
    const i8* arow = act + ((long)(b * H + ih) * W + iw) * Cin;
    const long kbase = (long)kk * Cin;

    for (int c0 = 0; c0 < Cin; c0 += 64) {
      v8i a = {0, 0, 0, 0, 0, 0, 0, 0};
      if (inb) {
        const i8* ap = arow + c0;
        const int2 p0 = *(const int2*)(ap +  0 + hi * 16);
        const int2 p1 = *(const int2*)(ap +  8 + hi * 16);
        const int2 p2 = *(const int2*)(ap + 32 + hi * 16);
        const int2 p3 = *(const int2*)(ap + 40 + hi * 16);
        a[0] = p0.x; a[1] = p0.y; a[2] = p1.x; a[3] = p1.y;
        a[4] = p2.x; a[5] = p2.y; a[6] = p3.x; a[7] = p3.y;
      }

      const long kidx = kbase + c0;
      __builtin_prefetch(brow + kidx + 64, 0, 1);

      #pragma unroll
      for (int j = 0; j < 4; ++j) {
        const i8* bp = brow + (long)j * 16 * Kt + kidx;
        const int4 q0 = *(const int4*)(bp + hi * 16);
        const int4 q1 = *(const int4*)(bp + 32 + hi * 16);
        v8i bf;
        bf[0] = q0.x; bf[1] = q0.y; bf[2] = q0.z; bf[3] = q0.w;
        bf[4] = q1.x; bf[5] = q1.y; bf[6] = q1.z; bf[7] = q1.w;
        acc[j] = __builtin_amdgcn_wmma_i32_16x16x64_iu8(true, a, true, bf,
                                                        acc[j], false, false);
      }
    }
  }

  #pragma unroll
  for (int j = 0; j < 4; ++j) {
    const int nCol = nT * 64 + j * 16 + r;
    #pragma unroll
    for (int v = 0; v < 8; ++v) {
      int mRow = mT * 16 + hi * 8 + v;
      if (mRow < M) out[(long)mRow * Cout + nCol] = (float)acc[j][v];
    }
  }
}

__global__ __launch_bounds__(256) void reduce_stats(const float* __restrict__ y,
                                                    long M, int C,
                                                    float* __restrict__ stats) {
  const int c = blockIdx.x;
  float s = 0.f, sq = 0.f;
  for (long m = threadIdx.x; m < M; m += 256) {
    float v = y[m * C + c];
    s += v; sq += v * v;
  }
  __shared__ float ls[256], lq[256];
  ls[threadIdx.x] = s; lq[threadIdx.x] = sq;
  __syncthreads();
  for (int off = 128; off > 0; off >>= 1) {
    if (threadIdx.x < (unsigned)off) {
      ls[threadIdx.x] += ls[threadIdx.x + off];
      lq[threadIdx.x] += lq[threadIdx.x + off];
    }
    __syncthreads();
  }
  if (threadIdx.x == 0) { stats[c] = ls[0]; stats[C + c] = lq[0]; }
}

__global__ __launch_bounds__(256) void bn_apply(const float* __restrict__ y,
                                                const float* __restrict__ stats,
                                                const float* __restrict__ gamma,
                                                const float* __restrict__ beta,
                                                long M, int C, int mode,
                                                i8* outB, float* outF, _Float16* outH) {
  long idx = (long)blockIdx.x * 256 + threadIdx.x;
  if (idx >= M * C) return;
  int c = (int)(idx % C);
  float mean = stats[c] / (float)M;
  float var  = stats[C + c] / (float)M - mean * mean;
  float v = (y[idx] - mean) * rsqrtf(var + BN_EPS) * gamma[c] + beta[c];
  if (mode == 0)      outB[idx] = (v >= 0.f) ? (i8)1 : (i8)-1;
  else if (mode == 1) outF[idx] = v;
  else                outH[idx] = (_Float16)fmaxf(v, 0.f);
}

__global__ __launch_bounds__(256) void maxpool_bin(const float* __restrict__ y,
                                                   i8* __restrict__ o) {
  long idx = (long)blockIdx.x * 256 + threadIdx.x;
  const long total = 8L * 19 * 19 * 512;
  if (idx >= total) return;
  int c = (int)(idx & 511);
  long t = idx >> 9;
  int w = (int)(t % 19); t /= 19;
  int h = (int)(t % 19);
  int b = (int)(t / 19);
  float mx = -3.4e38f;
  for (int dh = -1; dh <= 1; ++dh) {
    int hh = h + dh; if (hh < 0 || hh >= 19) continue;
    for (int dw = -1; dw <= 1; ++dw) {
      int ww = w + dw; if (ww < 0 || ww >= 19) continue;
      mx = fmaxf(mx, y[(((long)b * 19 + hh) * 19 + ww) * 512 + c]);
    }
  }
  o[idx] = (mx >= 0.f) ? (i8)1 : (i8)-1;
}

__global__ __launch_bounds__(256) void f32_to_f16(const float* __restrict__ a,
                                                  _Float16* __restrict__ o, long n) {
  long idx = (long)blockIdx.x * 256 + threadIdx.x;
  if (idx < n) o[idx] = (_Float16)a[idx];
}

__global__ __launch_bounds__(256) void gemm_f16_wmma(const _Float16* __restrict__ A,
                                                     const _Float16* __restrict__ Bw,
                                                     float* __restrict__ out,
                                                     int Mp, int N, int K) {
  const int lane = threadIdx.x & 31;
  const int wave = threadIdx.x >> 5;
  const int nTiles = N >> 6;
  const long tile = (long)blockIdx.x * 8 + wave;
  if (tile >= (long)(Mp >> 4) * nTiles) return;
  const int nT = (int)(tile % nTiles);
  const int mT = (int)(tile / nTiles);
  const int hi = lane >> 4;
  const int r  = lane & 15;

  const char* arow = (const char*)(A  + (long)(mT * 16 + r) * K);
  const char* brow = (const char*)(Bw + (long)(nT * 64 + r) * K);
  const long rowB = 16L * K * 2;

  v8f acc[4];
  #pragma unroll
  for (int j = 0; j < 4; ++j) acc[j] = (v8f){0.f,0.f,0.f,0.f,0.f,0.f,0.f,0.f};

  for (int k0 = 0; k0 < K; k0 += 32) {
    v8h a0 = *(const v8h*)(arow + k0 * 2 + hi * 32);
    v8h a1 = *(const v8h*)(arow + k0 * 2 + hi * 32 + 16);
    v16h av = __builtin_shufflevector(a0, a1, 0,1,2,3,4,5,6,7,8,9,10,11,12,13,14,15);
    __builtin_prefetch(brow + k0 * 2 + 64, 0, 1);
    #pragma unroll
    for (int j = 0; j < 4; ++j) {
      const char* bp = brow + (long)j * rowB + k0 * 2 + hi * 32;
      v8h b0 = *(const v8h*)(bp);
      v8h b1 = *(const v8h*)(bp + 16);
      v16h bv = __builtin_shufflevector(b0, b1, 0,1,2,3,4,5,6,7,8,9,10,11,12,13,14,15);
      acc[j] = __builtin_amdgcn_wmma_f32_16x16x32_f16(false, av, false, bv,
                                                      (short)0, acc[j], false, false);
    }
  }
  #pragma unroll
  for (int j = 0; j < 4; ++j) {
    const int nCol = nT * 64 + j * 16 + r;
    #pragma unroll
    for (int v = 0; v < 8; ++v)
      out[(long)(mT * 16 + hi * 8 + v) * N + nCol] = acc[j][v];
  }
}

__global__ __launch_bounds__(256) void bn_relu_nchw(const float* __restrict__ y,
                                                    const float* __restrict__ stats,
                                                    const float* __restrict__ gamma,
                                                    const float* __restrict__ beta,
                                                    float* __restrict__ out) {
  long idx = (long)blockIdx.x * 256 + threadIdx.x;
  const long total = 8L * 1024 * 361;
  if (idx >= total) return;
  int s = (int)(idx % 361);
  long t = idx / 361;
  int c = (int)(t % 1024);
  int b = (int)(t / 1024);
  const float Mf = 2888.f;
  float mean = stats[c] / Mf;
  float var  = stats[1024 + c] / Mf - mean * mean;
  float v = (y[((long)b * 361 + s) * 1024 + c] - mean) * rsqrtf(var + BN_EPS)
            * gamma[c] + beta[c];
  out[idx] = fmaxf(v, 0.f);
}

struct Spec { int cin, cout, stride, H, W; };
static const Spec specs[17] = {
  {3,64,1,300,300},{64,64,1,300,300},{64,128,2,300,300},{128,128,1,150,150},
  {128,128,1,150,150},{128,256,2,150,150},{256,256,1,75,75},{256,256,1,75,75},
  {256,256,1,75,75},{256,512,2,75,75},{512,512,1,38,38},{512,512,1,38,38},
  {512,512,1,38,38},{512,512,2,38,38},{512,512,1,19,19},{512,512,1,19,19},
  {512,512,1,19,19}
};

static inline long cdiv(long a, long b) { return (a + b - 1) / b; }

#define OFF_F   0L
#define OFF_I   184320000L
#define OFF_W8  230400000L
#define OFF_ST  235118592L
#define OFF_HA  235126784L
#define OFF_HW  241057792L
#define OFF_F2  243154944L

extern "C" void kernel_launch(void* const* d_in, const int* in_sizes, int n_in,
                              void* d_out, int out_size, void* d_ws, size_t ws_size,
                              hipStream_t stream) {
  (void)in_sizes; (void)n_in; (void)out_size; (void)ws_size;
  char* ws = (char*)d_ws;
  float*    F  = (float*)(ws + OFF_F);
  i8*       I  = (i8*)(ws + OFF_I);
  i8*       W8 = (i8*)(ws + OFF_W8);
  float*    ST = (float*)(ws + OFF_ST);
  _Float16* HA = (_Float16*)(ws + OFF_HA);
  _Float16* HW = (_Float16*)(ws + OFF_HW);
  float*    F2 = (float*)(ws + OFF_F2);

  const float* x = (const float*)d_in[0];

  {
    long M = 8L * 300 * 300;
    conv1_f32<<<cdiv(M * 64, 256), 256, 0, stream>>>(x, (const float*)d_in[1], F);
    reduce_stats<<<64, 256, 0, stream>>>(F, M, 64, ST);
    bn_apply<<<cdiv(M * 64, 256), 256, 0, stream>>>(
        F, ST, (const float*)d_in[2], (const float*)d_in[3],
        M, 64, 0, I, nullptr, nullptr);
  }

  for (int l = 1; l < 17; ++l) {
    const Spec s = specs[l];
    const int OH = (s.stride == 2) ? ((s.H - 1) / 2 + 1) : s.H;
    const int OW = (s.stride == 2) ? ((s.W - 1) / 2 + 1) : s.W;
    const long M = 8L * OH * OW;
    binarize_w<<<cdiv((long)s.cout * 9 * s.cin, 256), 256, 0, stream>>>(
        (const float*)d_in[1 + 3 * l], W8, s.cout, s.cin);
    const long tiles = cdiv(M, 16) * (s.cout / 64);
    conv_bin_wmma<<<cdiv(tiles, 8), 256, 0, stream>>>(
        I, W8, F, s.H, s.W, OH, OW, s.cin, s.cout, s.stride, 1);
    reduce_stats<<<s.cout, 256, 0, stream>>>(F, M, s.cout, ST);
    const int mode = (l == 16) ? 1 : 0;
    bn_apply<<<cdiv(M * s.cout, 256), 256, 0, stream>>>(
        F, ST, (const float*)d_in[2 + 3 * l], (const float*)d_in[3 + 3 * l],
        M, s.cout, mode, I, F2, nullptr);
  }

  maxpool_bin<<<cdiv(8L * 19 * 19 * 512, 256), 256, 0, stream>>>(F2, I);

  {
    const long M = 8L * 19 * 19;
    binarize_w<<<cdiv(1024L * 9 * 512, 256), 256, 0, stream>>>(
        (const float*)d_in[52], W8, 1024, 512);
    const long tiles = cdiv(M, 16) * (1024 / 64);
    conv_bin_wmma<<<cdiv(tiles, 8), 256, 0, stream>>>(
        I, W8, F, 19, 19, 19, 19, 512, 1024, 1, 6);
    reduce_stats<<<1024, 256, 0, stream>>>(F, M, 1024, ST);
    hipMemsetAsync(HA, 0, 2896L * 1024 * sizeof(_Float16), stream);
    bn_apply<<<cdiv(M * 1024, 256), 256, 0, stream>>>(
        F, ST, (const float*)d_in[53], (const float*)d_in[54],
        M, 1024, 2, nullptr, nullptr, HA);
  }

  {
    f32_to_f16<<<cdiv(1024L * 1024, 256), 256, 0, stream>>>(
        (const float*)d_in[55], HW, 1024L * 1024);
    const int Mp = 2896;
    const long tiles = (long)(Mp / 16) * (1024 / 64);
    gemm_f16_wmma<<<cdiv(tiles, 8), 256, 0, stream>>>(HA, HW, F2, Mp, 1024, 1024);
    reduce_stats<<<1024, 256, 0, stream>>>(F2, 2888, 1024, ST);
    bn_relu_nchw<<<cdiv(8L * 1024 * 361, 256), 256, 0, stream>>>(
        F2, ST, (const float*)d_in[57], (const float*)d_in[58], (float*)d_out);
  }
}
